// PixelContrastLoss_36429912605174
// MI455X (gfx1250) — matrix operand order repaired and checked by reading, not compiled, not run on hardware
//
#include <hip/hip_runtime.h>
#include <hip/hip_bf16.h>
#include <math.h>

// ---------------------------------------------------------------------------
// PixelContrastLoss for MI455X (gfx1250, wave32, WMMA + async-to-LDS)
//   feats  : [4, 256, 128, 128] f32
//   labels : [4, 128, 128] i32
//   predict: [4, 128, 128] i32
//   out    : scalar f32
// ---------------------------------------------------------------------------

typedef __attribute__((ext_vector_type(16))) _Float16 v16h;
typedef __attribute__((ext_vector_type(8)))  float    v8f;

#define PCL_B     4
#define PCL_C     19
#define PCL_D     256
#define PCL_P     16384          // 128*128
#define PCL_V     50             // n_view
#define PCL_T     (PCL_B*PCL_C)  // 76
#define PCL_N     (PCL_T*PCL_V)  // 3800
#define PCL_NPAD  3840           // padded rows (multiple of 128)
#define PCL_NTILE 238            // ceil(3800/16)
#define PCL_INVT  10.0f          // 1/temperature

// ---------------------------------------------------------------------------
// 1) Deterministic anchor sampling: one wave per (image,class).
//    Ballot/prefix-popcount gives a deterministic ordering of hard/easy
//    member indices (no atomics). Applies the reference's nh_keep rule with
//    a hash-derived x in [0,50).
// ---------------------------------------------------------------------------
__global__ void __launch_bounds__(32)
pcl_sample_kernel(const int* __restrict__ labels,
                  const int* __restrict__ predict,
                  int* __restrict__ sel /* [76][50] */) {
  int t = blockIdx.x;
  int b = t / PCL_C, c = t % PCL_C;
  int lane = threadIdx.x;
  __shared__ int hlist[64];
  __shared__ int elist[64];
  unsigned nh = 0, ne = 0;
  unsigned ltmask = (1u << lane) - 1u;
  const int* lb = labels  + b * PCL_P;
  const int* pb = predict + b * PCL_P;
  for (int base = 0; base < PCL_P; base += 32) {
    int p  = base + lane;
    int lv = lb[p], pv = pb[p];
    bool hard = (lv == c) && (pv != c);
    bool easy = (lv == c) && (pv == c);
    unsigned hb = (unsigned)__ballot(hard);
    unsigned eb = (unsigned)__ballot(easy);
    if (hard) { unsigned pos = nh + __popc(hb & ltmask); if (pos < 64u) hlist[pos] = p; }
    if (easy) { unsigned pos = ne + __popc(eb & ltmask); if (pos < 64u) elist[pos] = p; }
    nh += __popc(hb);
    ne += __popc(eb);
  }
  __syncthreads();
  unsigned nhl = nh < 64u ? nh : 64u;
  unsigned nel = ne < 64u ? ne : 64u;
  // pseudo-random x (stands in for threefry randint)
  unsigned seed = (unsigned)(b * 1315423911u) ^ (unsigned)(c * 2654435761u) ^ 0x9e3779b9u;
  seed ^= seed >> 13; seed *= 0x85ebca6bu; seed ^= seed >> 16;
  int x = (int)(seed % PCL_V);
  int nh_keep;
  if ((int)nh >= x && (int)ne >= PCL_V - x) nh_keep = x;
  else if (PCL_V >= (int)nh)                nh_keep = (int)nh;
  else                                      nh_keep = PCL_V - (int)ne;
  if (nh_keep < 0)     nh_keep = 0;
  if (nh_keep > PCL_V) nh_keep = PCL_V;
  for (int j = lane; j < PCL_V; j += 32) {
    int p;
    if (j < nh_keep) {
      p = ((unsigned)j < nhl) ? hlist[j] : j;
    } else {
      int e = j - nh_keep;
      p = ((unsigned)e < nel) ? elist[e] : j;
    }
    sel[t * PCL_V + j] = p;
  }
}

// ---------------------------------------------------------------------------
// 2) Gather sampled features into a row-major f16 matrix cf [NPAD][256].
//    Row r = k*T + t  (matches torch.cat(torch.unbind(feats_, 1), 0)).
//    Pad rows are zeroed.
// ---------------------------------------------------------------------------
__global__ void __launch_bounds__(256)
pcl_gather_kernel(const float* __restrict__ feats,
                  const int* __restrict__ sel,
                  _Float16* __restrict__ cfh) {
  int r = blockIdx.x;
  int d = threadIdx.x;
  if (r < PCL_N) {
    int t = r % PCL_T, k = r / PCL_T;
    int b = t / PCL_C;
    int pix = sel[t * PCL_V + k];
    float v = feats[((size_t)b * PCL_D + d) * PCL_P + pix];
    cfh[(size_t)r * PCL_D + d] = (_Float16)v;
  } else {
    cfh[(size_t)r * PCL_D + d] = (_Float16)0.0f;
  }
}

// ---------------------------------------------------------------------------
// 3) WMMA recompute passes over S = (cf @ cf^T) / temperature.
//    PASS 1: rowmax_i  = max_j S_ij           (j < N)
//    PASS 2: neg_i     = sum_j [lab_j!=lab_i] exp(S_ij - rowmax_i)
//    PASS 3: possum_i  = sum_j [lab_j==lab_i, j!=i]
//                          (v - log(exp(v) + neg_i + 1e-6)),  v = S_ij - rowmax_i
//    Block = 128 threads (4 waves); each wave owns a 16-row strip and keeps
//    its 8 A-fragments (16x256 f16) in VGPRs for the whole column sweep.
//    B tiles (16 contiguous cf rows = 8 KB) are staged with double-buffered
//    GLOBAL_LOAD_ASYNC_TO_LDS_B128 (ASYNCcnt), overlapping copy of tile jt+1
//    with the 8-WMMA compute of tile jt; one barrier per tile.
// ---------------------------------------------------------------------------
template <int PASS>
__global__ void __launch_bounds__(128)
pcl_pass_kernel(const _Float16* __restrict__ cfh,
                const float* __restrict__ rowmax,
                const float* __restrict__ negsum,
                float* __restrict__ outbuf) {
  __shared__ __align__(16) _Float16 Btile[2][16 * PCL_D];   // 2 x 8 KB
  int tid  = threadIdx.x;
  int wave = tid >> 5, lane = tid & 31;
  int hi   = lane >> 4, nn = lane & 15;
  int rowbase = blockIdx.x * 64 + wave * 16;

  // A fragments: lane = row (rowbase + nn); per K-step two contiguous 16B
  // chunks selected by hi per the 16-bit 16x32 A layout.
  v16h afrag[8];
  const _Float16* arow = cfh + (size_t)(rowbase + nn) * PCL_D;
  #pragma unroll
  for (int s = 0; s < 8; ++s) {
    int off = s * 32 + hi * 8;
    ((uint4*)&afrag[s])[0] = *(const uint4*)(arow + off);
    ((uint4*)&afrag[s])[1] = *(const uint4*)(arow + off + 16);
  }

  int   gr[8]; int li[8];
  float rm[8], ng[8], acc[8];
  #pragma unroll
  for (int r = 0; r < 8; ++r) {
    gr[r]  = rowbase + r + 8 * hi;              // C/D layout: M = r + 8*(lane>>4)
    li[r]  = (gr[r] % PCL_T) % PCL_C;
    acc[r] = (PASS == 1) ? -3.4e38f : 0.0f;
    rm[r]  = (PASS >= 2) ? rowmax[gr[r]] : 0.0f;
    ng[r]  = (PASS == 3) ? negsum[gr[r]] : 0.0f;
  }

  // Async B-tile staging: 128 lanes x 4 x b128 = 8 KB per tile.
  // LDS destination = low 32 bits of the generic shared-memory address
  // (HW maps flat LDS-aperture addresses via addr[31:0]).
  unsigned lds0 = (unsigned)(uintptr_t)(&Btile[0][0]) + (unsigned)tid * 16u;
  unsigned long long gbase =
      (unsigned long long)(uintptr_t)cfh + (unsigned long long)tid * 16ull;

  auto issue_tile = [&](int jt, int buf) {
    unsigned ldst = lds0 + (unsigned)buf * (16u * PCL_D * 2u);
    unsigned long long ga = gbase + (unsigned long long)jt * (16ull * PCL_D * 2ull);
    #pragma unroll
    for (int it = 0; it < 4; ++it) {
      asm volatile("global_load_async_to_lds_b128 %0, %1, off"
                   :: "v"(ldst + (unsigned)(it * 128 * 16)),
                      "v"(ga + (unsigned long long)(it * 128 * 16))
                   : "memory");
    }
  };

  issue_tile(0, 0);
  for (int jt = 0; jt < PCL_NTILE; ++jt) {
    int cur = jt & 1;
    asm volatile("s_wait_asynccnt 0" ::: "memory");  // my async loads landed
    __syncthreads();                                 // everyone's landed; prev buf free
    if (jt + 1 < PCL_NTILE) issue_tile(jt + 1, cur ^ 1);

    v8f c = {};
    #pragma unroll
    for (int s = 0; s < 8; ++s) {
      // B fragment: lane holds column n = lane&15, K chunks by lane>>4:
      // 8 halves at Btile[cur][n*256 + s*32 + hi*8] and 8 more at +16
      v16h bfrag;
      const _Float16* bp = &Btile[cur][0] + nn * PCL_D + s * 32 + hi * 8;
      ((uint4*)&bfrag)[0] = *(const uint4*)(bp);
      ((uint4*)&bfrag)[1] = *(const uint4*)(bp + 16);
      c = __builtin_amdgcn_wmma_f32_16x16x32_f16(
              false, afrag[s], false, bfrag, (short)0, c, false, false);
    }

    int  j   = jt * 16 + nn;
    int  lj  = (j % PCL_T) % PCL_C;
    bool jin = (j < PCL_N);
    #pragma unroll
    for (int r = 0; r < 8; ++r) {
      float l = c[r] * PCL_INVT;
      if (PASS == 1) {
        acc[r] = fmaxf(acc[r], jin ? l : -3.4e38f);
      } else if (PASS == 2) {
        if (jin && lj != li[r]) acc[r] += __expf(l - rm[r]);
      } else {
        if (jin && lj == li[r] && j != gr[r]) {
          float v = l - rm[r];
          acc[r] += v - __logf(__expf(v) + ng[r] + 1e-6f);
        }
      }
    }
  }

  // reduce across the 16 lanes sharing each row (xor masks < 16 keep hi group)
  #pragma unroll
  for (int r = 0; r < 8; ++r) {
    float a = acc[r];
    #pragma unroll
    for (int m = 1; m < 16; m <<= 1) {
      float o = __shfl_xor(a, m);
      a = (PASS == 1) ? fmaxf(a, o) : (a + o);
    }
    if (nn == 0) outbuf[gr[r]] = a;
  }
}

// ---------------------------------------------------------------------------
// 4) Final reduction: loss = -T * mean_i( possum_i / (B*V - 1) )
// ---------------------------------------------------------------------------
__global__ void __launch_bounds__(256)
pcl_finalize_kernel(const float* __restrict__ possum, float* __restrict__ out) {
  __shared__ float red[256];
  float s = 0.0f;
  for (int i = threadIdx.x; i < PCL_N; i += 256) s += possum[i];
  red[threadIdx.x] = s;
  __syncthreads();
  for (int w = 128; w > 0; w >>= 1) {
    if (threadIdx.x < (unsigned)w) red[threadIdx.x] += red[threadIdx.x + w];
    __syncthreads();
  }
  if (threadIdx.x == 0)
    out[0] = -0.1f * (red[0] / (float)PCL_N) / (float)(PCL_B * PCL_V - 1);
}

// ---------------------------------------------------------------------------
extern "C" void kernel_launch(void* const* d_in, const int* in_sizes, int n_in,
                              void* d_out, int out_size, void* d_ws, size_t ws_size,
                              hipStream_t stream) {
  const float* feats   = (const float*)d_in[0];
  const int*   labels  = (const int*)d_in[1];
  const int*   predict = (const int*)d_in[2];
  float*       out     = (float*)d_out;
  char*        ws      = (char*)d_ws;

  // workspace layout (256B-aligned slices), total ~2.03 MB
  int*      sel    = (int*)(ws);                 // 76*50*4   = 15200 B
  float*    rowmax = (float*)(ws + 16384);       // 3840*4    = 15360 B
  float*    negb   = (float*)(ws + 32768);       // 3840*4
  float*    possum = (float*)(ws + 49152);       // 3840*4
  _Float16* cfh    = (_Float16*)(ws + 65536);    // 3840*256*2 = 1966080 B
  (void)in_sizes; (void)n_in; (void)out_size; (void)ws_size;

  pcl_sample_kernel<<<PCL_T, 32, 0, stream>>>(labels, predict, sel);
  pcl_gather_kernel<<<PCL_NPAD, 256, 0, stream>>>(feats, sel, cfh);
  pcl_pass_kernel<1><<<PCL_NPAD / 64, 128, 0, stream>>>(cfh, rowmax, negb, rowmax);
  pcl_pass_kernel<2><<<PCL_NPAD / 64, 128, 0, stream>>>(cfh, rowmax, negb, negb);
  pcl_pass_kernel<3><<<PCL_NPAD / 64, 128, 0, stream>>>(cfh, rowmax, negb, possum);
  pcl_finalize_kernel<<<1, 256, 0, stream>>>(possum, out);
}
